// FineGrainedRetriever_37237366456561
// MI455X (gfx1250) — matrix operand order repaired and checked by reading, not compiled, not run on hardware
//
#include <hip/hip_runtime.h>
#include <hip/hip_bf16.h>

// MI455X (gfx1250) implementation: wave32, WMMA f32_16x16x32_f16 matrix path,
// f16 weights staged in LDS in *fragment-major* (swizzled) layout so every
// B-fragment is one contiguous 32B v16h load (2x ds_load_b128), f32 accum.
// All GEMMs tiled 128(M)x128(N), 8 waves/block, wave = 16-row strip.

typedef __attribute__((ext_vector_type(16))) _Float16 v16h;
typedef __attribute__((ext_vector_type(8)))  float    v8f;

#define EMB   128
#define GRP   8
#define TAUC  1.0f

// ---------- WMMA fragment helpers (ISA 7.12.2 16-bit layouts, wave32) ----------

__device__ inline v8f wmma_f16(v16h a, v16h b, v8f c) {
  return __builtin_amdgcn_wmma_f32_16x16x32_f16(false, a, false, b, (short)0, c,
                                                false, false);
}

// A-matrix 16x32 f16 fragment from an fp32 row pointer (row = lane&15).
// Element e<8 -> K = half8+e ; e>=8 -> K = 16+half8+(e-8). Two contiguous
// 8-float runs -> 4x global_load_b128.
__device__ inline v16h afrag_f32(const float* rowp, float scale) {
  const int half8 = ((threadIdx.x & 31) >> 4) * 8;
  const float4 q0 = *(const float4*)(rowp + half8);
  const float4 q1 = *(const float4*)(rowp + half8 + 4);
  const float4 q2 = *(const float4*)(rowp + 16 + half8);
  const float4 q3 = *(const float4*)(rowp + 16 + half8 + 4);
  v16h a;
  a[0]  = (_Float16)(q0.x * scale); a[1]  = (_Float16)(q0.y * scale);
  a[2]  = (_Float16)(q0.z * scale); a[3]  = (_Float16)(q0.w * scale);
  a[4]  = (_Float16)(q1.x * scale); a[5]  = (_Float16)(q1.y * scale);
  a[6]  = (_Float16)(q1.z * scale); a[7]  = (_Float16)(q1.w * scale);
  a[8]  = (_Float16)(q2.x * scale); a[9]  = (_Float16)(q2.y * scale);
  a[10] = (_Float16)(q2.z * scale); a[11] = (_Float16)(q2.w * scale);
  a[12] = (_Float16)(q3.x * scale); a[13] = (_Float16)(q3.y * scale);
  a[14] = (_Float16)(q3.z * scale); a[15] = (_Float16)(q3.w * scale);
  return a;
}

// ---- fragment-major (swizzled) LDS weight layout ----
// For B-tile (ktile: 32 K-rows, nt: 16 cols), lane = n + 16*khalf holds the
// 16 f16 values K = ktile*32 + khalf*8 + (0..7), ktile*32 + 16 + khalf*8 + (0..7) at one contiguous address:
//   idx = (((ktile*8 + nt)*32 + lane) << 4) + j
__device__ inline void stage_w_sw(const float* W, _Float16* shW, int krows) {
  const int elems = krows * EMB;
  for (int i = threadIdx.x; i < elems; i += blockDim.x) {  // i = k*128 + n
    const int k = i >> 7, n = i & 127;
    const int nt = n >> 4, nl = n & 15;
    const int ktile = k >> 5, ko = k & 31;
    const int lane = nl + (((ko >> 3) & 1) << 4);
    const int j = (ko & 7) + ((ko >> 4) << 3);
    shW[(((ktile * 8 + nt) * 32 + lane) << 4) + j] = (_Float16)W[i];
  }
}

__device__ inline v16h bfrag_sw(const _Float16* shW, int ktile, int nt) {
  return *(const v16h*)(shW +
      ((((ktile * 8 + nt) * 32 + (threadIdx.x & 31))) << 4));
}

// A-fragment-major LDS tile (for proj hidden): fragment for (row m, ktile)
// and lane-half h lives at ((m*4 + ktile)*2 + h) << 4, 16 contiguous f16.
__device__ inline v16h afrag_sw(const _Float16* shT, int m, int ktile) {
  const int h = (threadIdx.x & 31) >> 4;
  return *(const v16h*)(shT + (((m * 4 + ktile) * 2 + h) << 4));
}

// ---------------- small utility kernels ----------------

__global__ void k_zero(float* p, long long n) {
  long long i = (long long)blockIdx.x * blockDim.x + threadIdx.x;
  if (i < n) p[i] = 0.0f;
}

__global__ void k_deg(const int* srcv, const int* dstv, float* deg, int E) {
  int e = blockIdx.x * blockDim.x + threadIdx.x;
  if (e < E) {
    atomicAdd(&deg[dstv[e]], 1.0f);
    atomicAdd(&deg[srcv[e]], 1.0f);
  }
}

__global__ void k_invdeg(float* deg, int Nn) {
  int i = blockIdx.x * blockDim.x + threadIdx.x;
  if (i < Nn) {
    float d = deg[i];
    deg[i] = 1.0f / (d < 1.0f ? 1.0f : d);
  }
}

// msg = h[src2] + ea2 scattered into agg[dst2] with f32 global atomics.
__global__ void k_agg(const float* hin, int ld_in, const float* ea,
                      const float* ea_rev, const int* srcv, const int* dstv,
                      float* agg, int E) {
  long long t = (long long)blockIdx.x * blockDim.x + threadIdx.x;
  long long tot = (long long)2 * E * 32;
  if (t >= tot) return;
  int c0 = (int)(t & 31) * 4;
  long long eidx = t >> 5;
  int e = (int)(eidx % E);
  int rev = (int)(eidx / E);
  int sn = rev ? dstv[e] : srcv[e];
  int dn = rev ? srcv[e] : dstv[e];
  const float* eap = (rev ? ea_rev : ea) + (size_t)e * EMB + c0;
  const float* hp  = hin + (size_t)sn * ld_in + c0;
  float* ap        = agg + (size_t)dn * EMB + c0;
  const float4 hv = *(const float4*)hp;
  const float4 ev = *(const float4*)eap;
  atomicAdd(&ap[0], hv.x + ev.x);
  atomicAdd(&ap[1], hv.y + ev.y);
  atomicAdd(&ap[2], hv.z + ev.z);
  atomicAdd(&ap[3], hv.w + ev.w);
}

// ---------------- fused proj MLP2: out = relu(A@W1+b1)@W2+b2 ----------------

__global__ __launch_bounds__(256) void k_proj(const float* A, const float* w1,
                                              const float* b1, const float* w2,
                                              const float* b2, float* out, int M) {
  __shared__ _Float16 shW[EMB * EMB];   // 32 KB, swizzled; W1 then W2
  __shared__ _Float16 shT[EMB * EMB];   // 32 KB hidden tile, A-fragment-major
  const int wv   = threadIdx.x >> 5;
  const int lane = threadIdx.x & 31;
  const int mrow = lane & 15;
  const int row  = min(blockIdx.x * 128 + wv * 16 + mrow, M - 1);

  stage_w_sw(w1, shW, EMB);
  __syncthreads();

  v8f acc[8] = {};
  for (int kt = 0; kt < 4; ++kt) {
    v16h a = afrag_f32(A + (size_t)row * EMB + kt * 32, 1.0f);
#pragma unroll
    for (int nt = 0; nt < 8; ++nt)
      acc[nt] = wmma_f16(a, bfrag_sw(shW, kt, nt), acc[nt]);
  }
  // bias + relu -> hidden tile in A-fragment-major layout (f16)
  {
    const int n_lane = lane & 15;
    const int mo = (lane >> 4) * 8;
#pragma unroll
    for (int nt = 0; nt < 8; ++nt) {
      const float bn = b1[nt * 16 + n_lane];
      const int n = nt * 16 + n_lane;     // this is K of the next GEMM
      const int ktile = n >> 5, ko = n & 31;
      const int h = (ko >> 3) & 1;
      const int e = (ko & 7) + ((ko >> 4) & 1) * 8;
#pragma unroll
      for (int r = 0; r < 8; ++r) {
        float v = acc[nt][r] + bn;
        const int m = wv * 16 + mo + r;
        shT[(((m * 4 + ktile) * 2 + h) << 4) + e] =
            (_Float16)(v > 0.0f ? v : 0.0f);
      }
    }
  }
  __syncthreads();
  stage_w_sw(w2, shW, EMB);
  __syncthreads();

  v8f acc2[8] = {};
  for (int kt = 0; kt < 4; ++kt) {
    v16h a = afrag_sw(shT, wv * 16 + mrow, kt);
#pragma unroll
    for (int nt = 0; nt < 8; ++nt)
      acc2[nt] = wmma_f16(a, bfrag_sw(shW, kt, nt), acc2[nt]);
  }
  const int n_lane = lane & 15;
  const int mo = (lane >> 4) * 8;
#pragma unroll
  for (int nt = 0; nt < 8; ++nt) {
    const float bn = b2[nt * 16 + n_lane];
#pragma unroll
    for (int r = 0; r < 8; ++r) {
      int mr = blockIdx.x * 128 + wv * 16 + mo + r;
      if (mr < M) out[(size_t)mr * EMB + nt * 16 + n_lane] = acc2[nt][r] + bn;
    }
  }
}

// -------- SAGE layer: h' = relu(h@Wself + (agg*invdeg)@Wneigh + b) --------

__global__ __launch_bounds__(256) void k_sage(const float* hin, int ld_in,
                                              const float* agg, const float* invdeg,
                                              const float* wself, const float* wneigh,
                                              const float* bias, float* hout,
                                              int ld_out, int Nn) {
  __shared__ _Float16 shW[256 * EMB];   // 64 KB swizzled: ktiles 0-3 Wself, 4-7 Wneigh
  stage_w_sw(wself, shW, EMB);
  stage_w_sw(wneigh, shW + 4 * 8 * 32 * 16, EMB);
  __syncthreads();
  const int wv   = threadIdx.x >> 5;
  const int lane = threadIdx.x & 31;
  const int row  = min(blockIdx.x * 128 + wv * 16 + (lane & 15), Nn - 1);
  const float sc = invdeg[row];

  v8f acc[8] = {};
  for (int kt = 0; kt < 4; ++kt) {
    v16h a = afrag_f32(hin + (size_t)row * ld_in + kt * 32, 1.0f);
#pragma unroll
    for (int nt = 0; nt < 8; ++nt)
      acc[nt] = wmma_f16(a, bfrag_sw(shW, kt, nt), acc[nt]);
  }
  for (int kt = 0; kt < 4; ++kt) {
    v16h a = afrag_f32(agg + (size_t)row * EMB + kt * 32, sc);
#pragma unroll
    for (int nt = 0; nt < 8; ++nt)
      acc[nt] = wmma_f16(a, bfrag_sw(shW, 4 + kt, nt), acc[nt]);
  }
  const int n_lane = lane & 15;
  const int mo = (lane >> 4) * 8;
#pragma unroll
  for (int nt = 0; nt < 8; ++nt) {
    const float bn = bias[nt * 16 + n_lane];
#pragma unroll
    for (int r = 0; r < 8; ++r) {
      int mr = blockIdx.x * 128 + wv * 16 + mo + r;
      if (mr < Nn) {
        float v = acc[nt][r] + bn;
        hout[(size_t)mr * ld_out + nt * 16 + n_lane] = v > 0.0f ? v : 0.0f;
      }
    }
  }
}

// -------- pred MLP with fused h_triple gather (K = 1024), layer2 = dot --------

__global__ __launch_bounds__(256) void k_pred(const float* q, const float* ea,
                                              const float* h_e, const int* srcv,
                                              const int* dstv, const float* w1,
                                              const float* b1, const float* w2,
                                              const float* b2, float* logits,
                                              int Me) {
  __shared__ _Float16 shW[256 * EMB];   // 64 KB chunk of W1 (256 K-rows), swizzled
  const int wv   = threadIdx.x >> 5;
  const int lane = threadIdx.x & 31;
  const int row  = min(blockIdx.x * 128 + wv * 16 + (lane & 15), Me - 1);
  const int s = srcv[row], d = dstv[row];

  v8f acc[8] = {};
  for (int kc = 0; kc < 1024; kc += 256) {
    __syncthreads();
    stage_w_sw(w1 + (size_t)kc * EMB, shW, 256);
    __syncthreads();
    for (int kt = 0; kt < 8; ++kt) {
      const int kb = kc + kt * 32;
      const float* rowp;
      if (kb < 128)      rowp = q   + (size_t)row * EMB + kb;          // q embds
      else if (kb < 512) rowp = h_e + (size_t)s * 384 + (kb - 128);    // h_e[src]
      else if (kb < 640) rowp = ea  + (size_t)row * EMB + (kb - 512);  // edge_attr
      else               rowp = h_e + (size_t)d * 384 + (kb - 640);    // h_e[dst]
      v16h a = afrag_f32(rowp, 1.0f);
#pragma unroll
      for (int nt = 0; nt < 8; ++nt)
        acc[nt] = wmma_f16(a, bfrag_sw(shW, kt, nt), acc[nt]);
    }
  }
  // layer 2: logit = relu(hidden) . w2 + b2, cross-lane reduce over N
  const int n_lane = lane & 15;
  float partial[8];
#pragma unroll
  for (int r = 0; r < 8; ++r) partial[r] = 0.0f;
#pragma unroll
  for (int nt = 0; nt < 8; ++nt) {
    const float bn = b1[nt * 16 + n_lane];
    const float wn = w2[nt * 16 + n_lane];
#pragma unroll
    for (int r = 0; r < 8; ++r) {
      float v = acc[nt][r] + bn;
      partial[r] += (v > 0.0f ? v : 0.0f) * wn;
    }
  }
#pragma unroll
  for (int r = 0; r < 8; ++r)
#pragma unroll
    for (int m = 1; m < 16; m <<= 1)
      partial[r] += __shfl_xor(partial[r], m, 32);
  if (n_lane == 0) {
    const int mo = (lane >> 4) * 8;
#pragma unroll
    for (int r = 0; r < 8; ++r) {
      int mr = blockIdx.x * 128 + wv * 16 + mo + r;
      if (mr < Me) logits[mr] = partial[r] + b2[0];
    }
  }
}

// ---------------- Gumbel scores + per-group softmax stats ----------------

__device__ inline unsigned hash_u32(unsigned x) {
  x ^= 42u * 0x9E3779B9u;
  x *= 0x85EBCA6Bu; x ^= x >> 13;
  x *= 0xC2B2AE35u; x ^= x >> 16;
  return x;
}

__global__ __launch_bounds__(256) void k_scores(const float* logits, float* scores,
                                                float* gstats, int Ep) {
  __shared__ float red[256];
  const int g = blockIdx.x;
  const float* lg = logits + (size_t)g * Ep;
  float* sc = scores + (size_t)g * Ep;

  float mx_s = -1e30f, mx_l = -1e30f;
  for (int i = threadIdx.x; i < Ep; i += blockDim.x) {
    unsigned h = hash_u32((unsigned)(g * Ep + i));
    float u = (float)(h >> 8) * (1.0f / 16777216.0f);
    u = fminf(fmaxf(u, 1e-10f), 1.0f - 1e-10f);
    float gum = -__logf(-__logf(u));
    float s = (lg[i] + gum) / TAUC;
    sc[i] = s;
    mx_s = fmaxf(mx_s, s);
    mx_l = fmaxf(mx_l, lg[i] / TAUC);
  }
  red[threadIdx.x] = mx_s; __syncthreads();
  for (int st = 128; st > 0; st >>= 1) {
    if (threadIdx.x < st) red[threadIdx.x] = fmaxf(red[threadIdx.x], red[threadIdx.x + st]);
    __syncthreads();
  }
  mx_s = red[0]; __syncthreads();
  red[threadIdx.x] = mx_l; __syncthreads();
  for (int st = 128; st > 0; st >>= 1) {
    if (threadIdx.x < st) red[threadIdx.x] = fmaxf(red[threadIdx.x], red[threadIdx.x + st]);
    __syncthreads();
  }
  mx_l = red[0]; __syncthreads();

  float sum_s = 0.0f, sum_l = 0.0f;
  for (int i = threadIdx.x; i < Ep; i += blockDim.x) {
    sum_s += __expf(sc[i] - mx_s);
    sum_l += __expf(lg[i] / TAUC - mx_l);
  }
  red[threadIdx.x] = sum_s; __syncthreads();
  for (int st = 128; st > 0; st >>= 1) {
    if (threadIdx.x < st) red[threadIdx.x] += red[threadIdx.x + st];
    __syncthreads();
  }
  sum_s = red[0]; __syncthreads();
  red[threadIdx.x] = sum_l; __syncthreads();
  for (int st = 128; st > 0; st >>= 1) {
    if (threadIdx.x < st) red[threadIdx.x] += red[threadIdx.x + st];
    __syncthreads();
  }
  if (threadIdx.x == 0) {
    gstats[g * 4 + 0] = mx_s;
    gstats[g * 4 + 1] = sum_s;
    gstats[g * 4 + 2] = mx_l;
    gstats[g * 4 + 3] = red[0];
  }
}

// ------- rank-by-count selection: rank order == sorted-descending top-K -------

__global__ void k_rank(const float* logits, const float* scores, const float* gstats,
                       float* prob, float* mask, int* topi, float* logit_out,
                       int G, int Ep, int K) {
  long long gi = (long long)blockIdx.x * blockDim.x + threadIdx.x;
  if (gi >= (long long)G * Ep) return;
  int g = (int)(gi / Ep), i = (int)(gi - (long long)g * Ep);
  const float* sc = scores + (size_t)g * Ep;
  const float si = sc[i];
  int rank = 0;
  for (int j = 0; j < Ep; ++j) {
    if ((j & 63) == 0) __builtin_prefetch(sc + j + 512, 0, 0);
    float sj = sc[j];
    rank += (sj > si) || (sj == si && j < i);
  }
  const float mx_s = gstats[g * 4 + 0], sum_s = gstats[g * 4 + 1];
  const float mx_l = gstats[g * 4 + 2], sum_l = gstats[g * 4 + 3];
  const float soft = __expf(si - mx_s) / sum_s;
  const float lg = logits[(size_t)g * Ep + i];
  const float hard = rank < K ? 1.0f : 0.0f;
  mask[(size_t)g * Ep + i] = hard - soft + soft;  // straight-through forward
  logit_out[(size_t)g * Ep + i] = lg;
  if (rank < K) {
    topi[(size_t)g * K + rank] = i;
    prob[(size_t)g * K + rank] = __expf(lg / TAUC - mx_l) / sum_l;
  }
}

// ---------------- host launcher ----------------

extern "C" void kernel_launch(void* const* d_in, const int* in_sizes, int n_in,
                              void* d_out, int out_size, void* d_ws, size_t ws_size,
                              hipStream_t stream) {
  const float* x      = (const float*)d_in[0];
  const float* eattr  = (const float*)d_in[1];
  const float* qemb   = (const float*)d_in[2];
  const int*   eidx   = (const int*)d_in[3];
  const float* wself  = (const float*)d_in[6];
  const float* wneigh = (const float*)d_in[7];
  const float* sageb  = (const float*)d_in[8];
  const float* pw1    = (const float*)d_in[9];
  const float* pb1    = (const float*)d_in[10];
  const float* pw2    = (const float*)d_in[11];
  const float* pb2    = (const float*)d_in[12];
  const float* rw1    = (const float*)d_in[13];
  const float* rb1    = (const float*)d_in[14];
  const float* rw2    = (const float*)d_in[15];
  const float* rb2    = (const float*)d_in[16];

  const int N  = in_sizes[0] / EMB;
  const int E  = in_sizes[1] / EMB;
  const int G  = GRP;
  const int Ep = E / G;
  const int K  = (out_size - 2 * G * Ep) / (2 * G);  // epoch=10 -> r=0.6

  const int* srcv = eidx;
  const int* dstv = eidx + E;

  // workspace partition (all f32)
  float* ea_rev = (float*)d_ws;
  float* agg    = ea_rev + (size_t)E * EMB;
  float* h_e    = agg + (size_t)N * EMB;      // [N, 3*EMB]
  float* deg    = h_e + (size_t)N * 3 * EMB;
  float* logits = deg + N;
  float* scores = logits + E;
  float* gstats = scores + E;

  float* out      = (float*)d_out;
  float* out_prob = out;
  float* out_mask = out + (size_t)G * K;
  int*   out_topi = (int*)(out + (size_t)G * K + (size_t)G * Ep);
  float* out_lg   = out + 2 * (size_t)G * K + (size_t)G * Ep;

  const int T = 256;
  auto blks = [](long long n, int t) { return (unsigned)((n + t - 1) / t); };

  // degrees
  k_zero<<<blks(N, T), T, 0, stream>>>(deg, N);
  k_deg<<<blks(E, T), T, 0, stream>>>(srcv, dstv, deg, E);
  k_invdeg<<<blks(N, T), T, 0, stream>>>(deg, N);

  // reverse-edge attrs: ea_rev = MLP2(edge_attr)
  k_proj<<<blks(E, 128), T, 0, stream>>>(eattr, pw1, pb1, pw2, pb2, ea_rev, E);

  // SAGE layers
  for (int l = 0; l < 3; ++l) {
    const float* hin = (l == 0) ? x : (h_e + (size_t)(l - 1) * EMB);
    const int ld_in  = (l == 0) ? EMB : 3 * EMB;
    k_zero<<<blks((long long)N * EMB, T), T, 0, stream>>>(agg, (long long)N * EMB);
    k_agg<<<blks((long long)2 * E * 32, T), T, 0, stream>>>(
        hin, ld_in, eattr, ea_rev, srcv, dstv, agg, E);
    k_sage<<<blks(N, 128), T, 0, stream>>>(
        hin, ld_in, agg, deg,
        wself + (size_t)l * EMB * EMB, wneigh + (size_t)l * EMB * EMB,
        sageb + (size_t)l * EMB, h_e + (size_t)l * EMB, 3 * EMB, N);
  }

  // pred MLP over gathered h_triple
  k_pred<<<blks(E, 128), T, 0, stream>>>(qemb, eattr, h_e, srcv, dstv,
                                         rw1, rb1, rw2, rb2, logits, E);

  // gumbel top-k
  k_scores<<<G, T, 0, stream>>>(logits, scores, gstats, Ep);
  k_rank<<<blks((long long)G * Ep, T), T, 0, stream>>>(
      logits, scores, gstats, out_prob, out_mask, out_topi, out_lg, G, Ep, K);
}
